// CMHAtt_83468394430962
// MI455X (gfx1250) — matrix operand order repaired and checked by reading, not compiled, not run on hardware
//
#include <hip/hip_runtime.h>

#define HIDDEN 1024
#define HEADS  16
#define DK     64
#define SEQ    512
#define BATCH  8
#define GN     100

typedef __attribute__((ext_vector_type(16))) __bf16 v16bf;
typedef __attribute__((ext_vector_type(8)))  __bf16 v8bf;
typedef __attribute__((ext_vector_type(4)))  __bf16 v4bf;
typedef __attribute__((ext_vector_type(8)))  float  v8f;

union ABv { v16bf v; v8bf h[2]; };

__device__ __forceinline__ v8f wmma_bf16(const ABv& a, const ABv& b, v8f c) {
    return __builtin_amdgcn_wmma_f32_16x16x32_bf16(
        /*neg_a=*/false, a.v, /*neg_b=*/false, b.v,
        /*c_mod=*/(short)0, c, /*reuse_a=*/false, /*reuse_b=*/false);
}

// ---------------- f32 -> bf16 conversion (4 elems / thread) ----------------
__global__ void cvt_f32_to_bf16(const float* __restrict__ in,
                                __bf16* __restrict__ out, int n4) {
    int i = blockIdx.x * blockDim.x + threadIdx.x;
    if (i >= n4) return;
    float4 f = ((const float4*)in)[i];
    v4bf o;
    o[0] = (__bf16)f.x; o[1] = (__bf16)f.y; o[2] = (__bf16)f.z; o[3] = (__bf16)f.w;
    ((v4bf*)out)[i] = o;
}

// ------- register-blocked GEMM: 32x64 output per wave (2 M x 4 N tiles) ----
// out = A(MxK) * W(NxK)^T ; A row-major lda, W row-major ldb.
// A-fragment: lane m=L%16 holds K in [8h,8h+8) u [16+8h,16+8h+8)  (2x16B loads)
// B-fragment: lane n=L%16 holds K in [8h,8h+8) u [16+8h,16+8h+8)  (2x16B loads)
__device__ __forceinline__ void gemm_2x4(const __bf16* __restrict__ A,
                                         const __bf16* __restrict__ W,
                                         int m0, int n0, int lda, int ldb,
                                         int K, int l16, int half,
                                         v8f acc[2][4]) {
    const __bf16* ar0 = A + (size_t)(m0 + l16) * lda;
    const __bf16* ar1 = ar0 + (size_t)16 * lda;
    const __bf16* br0 = W + (size_t)(n0 + l16) * ldb;
    for (int k = 0; k < K; k += 32) {
        ABv a0, a1;
        a0.h[0] = *(const v8bf*)(ar0 + k + 8 * half);
        a0.h[1] = *(const v8bf*)(ar0 + k + 16 + 8 * half);
        a1.h[0] = *(const v8bf*)(ar1 + k + 8 * half);
        a1.h[1] = *(const v8bf*)(ar1 + k + 16 + 8 * half);
        #pragma unroll
        for (int nj = 0; nj < 4; ++nj) {
            ABv b;
            const __bf16* br = br0 + (size_t)(nj * 16) * ldb;
            b.h[0] = *(const v8bf*)(br + k + 8 * half);
            b.h[1] = *(const v8bf*)(br + k + 16 + 8 * half);
            acc[0][nj] = wmma_bf16(a0, b, acc[0][nj]);
            acc[1][nj] = wmma_bf16(a1, b, acc[1][nj]);
        }
    }
}

// ---------------- Q/K/V projections (z = 0/1/2), head-split store ----------
// N-group = 64 cols = exactly one head (HIDDEN/DK == 16 n-groups).
__global__ __launch_bounds__(256)
void proj_kernel(const __bf16* __restrict__ xq, const __bf16* __restrict__ xk,
                 const __bf16* __restrict__ xv,
                 const __bf16* __restrict__ wq, const __bf16* __restrict__ wk,
                 const __bf16* __restrict__ wv,
                 const float* __restrict__ bq, const float* __restrict__ bk,
                 const float* __restrict__ bv,
                 __bf16* __restrict__ qh, __bf16* __restrict__ kh,
                 __bf16* __restrict__ vt) {
    int wid  = threadIdx.x >> 5;
    int lane = threadIdx.x & 31, l16 = lane & 15, half = lane >> 4;
    int t  = blockIdx.x * 8 + wid;          // 2048 jobs: 128 (M/32) x 16 (N/64)
    int mg = t >> 4, ng = t & 15;
    int m0 = mg * 32, n0 = ng * 64;
    const __bf16 *X, *W; const float* bias;
    int z = blockIdx.z;
    if (z == 0)      { X = xq; W = wq; bias = bq; }
    else if (z == 1) { X = xk; W = wk; bias = bk; }
    else             { X = xv; W = wv; bias = bv; }

    v8f acc[2][4];
    #pragma unroll
    for (int mi = 0; mi < 2; ++mi)
        #pragma unroll
        for (int nj = 0; nj < 4; ++nj)
            acc[mi][nj] = (v8f){0.f,0.f,0.f,0.f,0.f,0.f,0.f,0.f};

    gemm_2x4(X, W, m0, n0, HIDDEN, HIDDEN, HIDDEN, l16, half, acc);

    int h = ng;                                 // head index
    #pragma unroll
    for (int nj = 0; nj < 4; ++nj) {
        int d = nj * 16 + l16;
        float bval = bias[n0 + d];
        #pragma unroll
        for (int mi = 0; mi < 2; ++mi) {
            #pragma unroll
            for (int i = 0; i < 8; ++i) {
                int row = m0 + mi * 16 + i + 8 * half;
                int b = row >> 9, s = row & 511;
                float val = acc[mi][nj][i] + bval;
                if (z < 2) {
                    __bf16* dst = (z == 0) ? qh : kh;   // (B,H,S,DK)
                    dst[(((size_t)(b * HEADS + h)) * SEQ + s) * DK + d] = (__bf16)val;
                } else {                                 // V^T: (B,H,DK,S)
                    vt[(((size_t)(b * HEADS + h)) * DK + d) * SEQ + s] = (__bf16)val;
                }
            }
        }
    }
}

// ---------------- attention: one wave per (b, h, 16-query tile) ------------
__global__ __launch_bounds__(32)
void attn_kernel(const __bf16* __restrict__ qh, const __bf16* __restrict__ kh,
                 const __bf16* __restrict__ vt, const float* __restrict__ graph,
                 const unsigned char* __restrict__ mask,
                 __bf16* __restrict__ attout) {
    __shared__ float  sS[16 * SEQ];   // f32 logits
    __shared__ __bf16 sP[16 * SEQ];   // bf16 exp(logits) == A-operand order
    __shared__ float  sInv[16];

    int lane = threadIdx.x & 31, l16 = lane & 15, half = lane >> 4;
    int bid = blockIdx.x;             // b*512 + h*32 + qt
    int qt = bid & 31, h = (bid >> 5) & 15, b = bid >> 9;
    size_t bh = (size_t)(b * HEADS + h);
    const __bf16* Q = qh + (bh * SEQ + (size_t)qt * 16) * DK;
    const __bf16* K = kh + bh * SEQ * DK;
    const __bf16* V = vt + bh * DK * SEQ;

    // Phase 1: logits = (Q K^T)/8 * g, masked, into LDS (f32)
    const __bf16* ar = Q + l16 * DK;
    for (int j = 0; j < 32; ++j) {
        const __bf16* br = K + (size_t)(j * 16 + l16) * DK;
        v8f c = {0.f,0.f,0.f,0.f,0.f,0.f,0.f,0.f};
        #pragma unroll
        for (int k = 0; k < DK; k += 32) {
            ABv a, bb;
            a.h[0]  = *(const v8bf*)(ar + k + 8 * half);
            a.h[1]  = *(const v8bf*)(ar + k + 16 + 8 * half);
            bb.h[0] = *(const v8bf*)(br + k + 8 * half);
            bb.h[1] = *(const v8bf*)(br + k + 16 + 8 * half);
            c = wmma_bf16(a, bb, c);
        }
        int colk = j * 16 + l16;
        bool mk = mask[b * SEQ + colk] != 0;
        #pragma unroll
        for (int i = 0; i < 8; ++i) {
            int rloc = i + 8 * half;
            int row  = qt * 16 + rloc;
            float s = c[i] * 0.125f;                       // 1/sqrt(64)
            float g = 1.f;
            if (row < GN && colk < GN)
                g = graph[(size_t)b * GN * GN + row * GN + colk] + (row == colk ? 1.f : 0.f);
            s *= g;
            if (mk) s = -1e9f;
            sS[rloc * SEQ + colk] = s;
        }
    }
    __syncthreads();

    // Phase 2: row softmax (row r handled by lanes r and r+16, half rows each)
    {
        int r = l16;
        const float* rowp = sS + r * SEQ + half * 256;
        float mx = -3.0e38f;
        for (int cidx = 0; cidx < 256; ++cidx) mx = fmaxf(mx, rowp[cidx]);
        mx = fmaxf(mx, __shfl_xor(mx, 16, 32));
        float sum = 0.f;
        __bf16* prow = sP + r * SEQ + half * 256;
        for (int cidx = 0; cidx < 256; ++cidx) {
            float e = __expf(rowp[cidx] - mx);
            sum += e;
            prow[cidx] = (__bf16)e;
        }
        sum += __shfl_xor(sum, 16, 32);
        if (half == 0) sInv[r] = 1.f / sum;
    }
    __syncthreads();

    // Phase 3: O = P * V  (P = A-operand from LDS, V^T rows = B columns)
    float inv[8];
    #pragma unroll
    for (int i = 0; i < 8; ++i) inv[i] = sInv[i + 8 * half];
    for (int nt = 0; nt < 4; ++nt) {
        v8f c = {0.f,0.f,0.f,0.f,0.f,0.f,0.f,0.f};
        const __bf16* br = V + (size_t)(nt * 16 + l16) * SEQ;
        #pragma unroll 4
        for (int k = 0; k < SEQ; k += 32) {
            ABv a, bb;
            a.h[0]  = *(const v8bf*)(sP + l16 * SEQ + k + 8 * half);
            a.h[1]  = *(const v8bf*)(sP + l16 * SEQ + k + 16 + 8 * half);
            bb.h[0] = *(const v8bf*)(br + k + 8 * half);
            bb.h[1] = *(const v8bf*)(br + k + 16 + 8 * half);
            c = wmma_bf16(a, bb, c);
        }
        #pragma unroll
        for (int i = 0; i < 8; ++i) {
            int s = qt * 16 + i + 8 * half;
            attout[((size_t)(b * SEQ + s)) * HIDDEN + h * DK + nt * 16 + l16] =
                (__bf16)(c[i] * inv[i]);
        }
    }
}

// ---------------- final linear: out = AttOut @ Wm^T + bm (f32 store) -------
__global__ __launch_bounds__(256)
void out_gemm(const __bf16* __restrict__ A, const __bf16* __restrict__ Wm,
              const float* __restrict__ bm, float* __restrict__ out) {
    int wid  = threadIdx.x >> 5;
    int lane = threadIdx.x & 31, l16 = lane & 15, half = lane >> 4;
    int t  = blockIdx.x * 8 + wid;          // 2048 jobs: 128 x 16
    int m0 = (t >> 4) * 32, n0 = (t & 15) * 64;

    v8f acc[2][4];
    #pragma unroll
    for (int mi = 0; mi < 2; ++mi)
        #pragma unroll
        for (int nj = 0; nj < 4; ++nj)
            acc[mi][nj] = (v8f){0.f,0.f,0.f,0.f,0.f,0.f,0.f,0.f};

    gemm_2x4(A, Wm, m0, n0, HIDDEN, HIDDEN, HIDDEN, l16, half, acc);

    #pragma unroll
    for (int nj = 0; nj < 4; ++nj) {
        int col = n0 + nj * 16 + l16;
        float bval = bm[col];
        #pragma unroll
        for (int mi = 0; mi < 2; ++mi) {
            #pragma unroll
            for (int i = 0; i < 8; ++i) {
                int row = m0 + mi * 16 + i + 8 * half;
                out[(size_t)row * HIDDEN + col] = acc[mi][nj][i] + bval;
            }
        }
    }
}

extern "C" void kernel_launch(void* const* d_in, const int* in_sizes, int n_in,
                              void* d_out, int out_size, void* d_ws, size_t ws_size,
                              hipStream_t stream) {
    (void)in_sizes; (void)n_in; (void)out_size; (void)ws_size;
    const float* v  = (const float*)d_in[0];
    const float* k  = (const float*)d_in[1];
    const float* q  = (const float*)d_in[2];
    const unsigned char* mask = (const unsigned char*)d_in[3];
    const float* graph = (const float*)d_in[4];
    const float* Wv = (const float*)d_in[5];
    const float* bv = (const float*)d_in[6];
    const float* Wk = (const float*)d_in[7];
    const float* bk = (const float*)d_in[8];
    const float* Wq = (const float*)d_in[9];
    const float* bq = (const float*)d_in[10];
    const float* Wm = (const float*)d_in[11];
    const float* bm = (const float*)d_in[12];
    float* out = (float*)d_out;

    const size_t ACT = (size_t)BATCH * SEQ * HIDDEN;   // 4 Mi elems
    const size_t WEL = (size_t)HIDDEN * HIDDEN;        // 1 Mi elems
    char* ws = (char*)d_ws;
    size_t off = 0;
    __bf16* xq  = (__bf16*)(ws + off); off += ACT * 2;
    __bf16* xk  = (__bf16*)(ws + off); off += ACT * 2;
    __bf16* xv  = (__bf16*)(ws + off); off += ACT * 2;
    __bf16* wqb = (__bf16*)(ws + off); off += WEL * 2;
    __bf16* wkb = (__bf16*)(ws + off); off += WEL * 2;
    __bf16* wvb = (__bf16*)(ws + off); off += WEL * 2;
    __bf16* wmb = (__bf16*)(ws + off); off += WEL * 2;
    __bf16* qhh = (__bf16*)(ws + off); off += ACT * 2;
    __bf16* khh = (__bf16*)(ws + off); off += ACT * 2;
    __bf16* vtt = (__bf16*)(ws + off); off += ACT * 2;
    __bf16* ao  = (__bf16*)(ws + off); off += ACT * 2;   // total 64 MB

    int nact4 = (int)(ACT / 4);   // 1048576
    int nw4   = (int)(WEL / 4);   // 262144
    cvt_f32_to_bf16<<<nact4 / 256, 256, 0, stream>>>(q,  xq,  nact4);
    cvt_f32_to_bf16<<<nact4 / 256, 256, 0, stream>>>(k,  xk,  nact4);
    cvt_f32_to_bf16<<<nact4 / 256, 256, 0, stream>>>(v,  xv,  nact4);
    cvt_f32_to_bf16<<<nw4   / 256, 256, 0, stream>>>(Wq, wqb, nw4);
    cvt_f32_to_bf16<<<nw4   / 256, 256, 0, stream>>>(Wk, wkb, nw4);
    cvt_f32_to_bf16<<<nw4   / 256, 256, 0, stream>>>(Wv, wvb, nw4);
    cvt_f32_to_bf16<<<nw4   / 256, 256, 0, stream>>>(Wm, wmb, nw4);

    proj_kernel<<<dim3(256, 1, 3), 256, 0, stream>>>(xq, xk, xv, wqb, wkb, wvb,
                                                     bq, bk, bv, qhh, khh, vtt);
    attn_kernel<<<BATCH * HEADS * 32, 32, 0, stream>>>(qhh, khh, vtt, graph, mask, ao);
    out_gemm<<<256, 256, 0, stream>>>(ao, wmb, bm, out);
}
